// AffinityNeuralNetworkCliffNetMONN_29300266893468
// MI455X (gfx1250) — matrix operand order repaired and checked by reading, not compiled, not run on hardware
//
#include <hip/hip_runtime.h>
#include <hip/hip_bf16.h>

#define BATCH 128
#define LCV   64
#define LPV   1024
#define HD    128
#define DLOOP 3

typedef __bf16 bf16;
typedef __attribute__((ext_vector_type(16))) __bf16 v16bf;
typedef __attribute__((ext_vector_type(8)))  float  v8f;

// ---------------------------------------------------------------------------
// Fast transcendentals: map straight onto v_exp_f32 / v_rcp_f32 instead of the
// branchy ocml tanh/div expansions (epilogues must co-execute with WMMA).
// ---------------------------------------------------------------------------
static __device__ inline float fast_tanh(float x) {
  // saturates to +/-1 for large |x| (exp->inf => rcp->0 ; exp->0 => 1-2*1=-1)
  const float e = __expf(2.f * x);
  return 1.f - 2.f * __builtin_amdgcn_rcpf(e + 1.f);
}
static __device__ inline float fast_sigmoid(float x) {
  return __builtin_amdgcn_rcpf(1.f + __expf(-x));
}

// ---------------------------------------------------------------------------
// WMMA helpers (wave32, 16x16x32 bf16 -> f32 accumulate)
// A: 16x32 row-major fragment. lane<16 -> row=lane, kbase=0 ; lane>=16 -> row=lane-16, kbase=8
//    element e: K = kbase + (e&7) + (e>=8 ? 16 : 0)
// B: 32x16 (KxN). lane col = lane&15, kbase = (lane>=16)?8:0 ; element e: K = kbase + (e&7) + (e>=8 ? 16 : 0)
// C/D: lane col = lane&15, rows mbase..mbase+7 with mbase = (lane>=16)?8:0
// ---------------------------------------------------------------------------
static __device__ inline v8f wmma_bf16(v16bf a, v16bf b, v8f c) {
  return __builtin_amdgcn_wmma_f32_16x16x32_bf16(false, a, false, b, (short)0, c,
                                                 false, false);
}

static __device__ inline v16bf loadA_f32(const float* __restrict__ A, int lda) {
  const int lane = threadIdx.x & 31;
  const float* r = A + (lane & 15) * lda + ((lane & 16) ? 8 : 0);
  v16bf a;
#pragma unroll
  for (int e = 0; e < 8; ++e) { a[e] = (bf16)r[e]; a[e + 8] = (bf16)r[e + 16]; }
  return a;
}

static __device__ inline v16bf loadA_bf(const bf16* __restrict__ A, int lda) {
  const int lane = threadIdx.x & 31;
  const bf16* r = A + (lane & 15) * lda + ((lane & 16) ? 8 : 0);
  v16bf a;
#pragma unroll
  for (int e = 0; e < 8; ++e) { a[e] = r[e]; a[e + 8] = r[e + 16]; }
  return a;
}

// B fragment from an N-major source: S[n*ld + k] (i.e. B^T layout, contiguous in k)
static __device__ inline v16bf loadB_nk(const bf16* __restrict__ S, int ld) {
  const int lane = threadIdx.x & 31;
  const bf16* r = S + (lane & 15) * ld + ((lane & 16) ? 8 : 0);
  v16bf b;
#pragma unroll
  for (int e = 0; e < 8; ++e) { b[e] = r[e]; b[e + 8] = r[e + 16]; }
  return b;
}

// ---------------------------------------------------------------------------
// Generic projection: Y = act(X @ W + bias), X:[N,128] f32, W:[128,128] f32
// (fan_in-major, W[k*128+n]).  One block = 64 rows.
// ACT: 1 = lrelu(0.1), 2 = tanh.  OUTF/OUTBF select the output stores.
// Compile-time specialization keeps the per-element epilogue branch-free.
// ---------------------------------------------------------------------------
template <int ACT, bool OUTF, bool OUTBF>
__global__ __launch_bounds__(256) void k_proj(const float* __restrict__ X,
                                              const float* __restrict__ W,
                                              const float* __restrict__ bias,
                                              float* __restrict__ Yf,
                                              bf16* __restrict__ Ybf) {
  __shared__ bf16 sW[HD * 136];  // W transposed: sW[n*136 + k]
  __shared__ bf16 sX[64 * 136];  // sX[r*136 + k]
  const int tid = threadIdx.x;
  const long row0 = (long)blockIdx.x * 64;
  for (int idx = tid; idx < HD * HD; idx += 256) {
    const int k = idx >> 7, n = idx & 127;
    sW[n * 136 + k] = (bf16)W[idx];
  }
  for (int idx = tid; idx < 64 * HD; idx += 256) {
    const int r = idx >> 7, k = idx & 127;
    sX[r * 136 + k] = (bf16)X[row0 * HD + idx];
  }
  const int wave = tid >> 5;
  const int rowTile = wave & 3;
  const int cbase = (wave >> 2) * 4;
  const int lane = tid & 31;
  const int lc = lane & 15;
  const int mb = (lane & 16) ? 8 : 0;
  // hoist uniform epilogue operands ahead of the WMMA groups
  float bv[4];
#pragma unroll
  for (int t = 0; t < 4; ++t) bv[t] = bias[(cbase + t) * 16 + lc];
  __syncthreads();
  const bf16* ar = sX + rowTile * 16 * 136;
  const v16bf a0 = loadA_bf(ar + 0, 136);
  const v16bf a1 = loadA_bf(ar + 32, 136);
  const v16bf a2 = loadA_bf(ar + 64, 136);
  const v16bf a3 = loadA_bf(ar + 96, 136);
#pragma unroll
  for (int t = 0; t < 4; ++t) {
    const int ct = cbase + t;
    const bf16* br = sW + ct * 16 * 136;
    v8f acc = {};
    acc = wmma_bf16(a0, loadB_nk(br + 0, 136), acc);
    acc = wmma_bf16(a1, loadB_nk(br + 32, 136), acc);
    acc = wmma_bf16(a2, loadB_nk(br + 64, 136), acc);
    acc = wmma_bf16(a3, loadB_nk(br + 96, 136), acc);
    const int col = ct * 16 + lc;
#pragma unroll
    for (int v = 0; v < 8; ++v) {
      float x = acc[v] + bv[t];
      if (ACT == 1) x = (x > 0.f) ? x : 0.1f * x;
      else if (ACT == 2) x = fast_tanh(x);
      const long o = (row0 + rowTile * 16 + mb + v) * HD + col;
      if (OUTF) Yf[o] = x;
      if (OUTBF) Ybf[o] = (bf16)x;
    }
  }
}

// ---------------------------------------------------------------------------
// pw = sigmoid(pc @ pp^T) per batch; stores pw [b,LC,LP] and pwT [b,LP,LC] (bf16)
// ---------------------------------------------------------------------------
__global__ __launch_bounds__(256) void k_pairwise(const bf16* __restrict__ pc,
                                                  const bf16* __restrict__ pp,
                                                  bf16* __restrict__ pw,
                                                  bf16* __restrict__ pwT) {
  const int b = blockIdx.x;
  __shared__ bf16 sA[LCV * 136];
  const int tid = threadIdx.x;
  const bf16* pcb = pc + (size_t)b * LCV * HD;
  const bf16* ppb = pp + (size_t)b * LPV * HD;
  for (int idx = tid; idx < LCV * HD; idx += 256) {
    const int r = idx >> 7, k = idx & 127;
    sA[r * 136 + k] = pcb[idx];
  }
  __syncthreads();
  const int wave = tid >> 5;
  const int rowTile = wave & 3;
  const int cg = wave >> 2;  // 0..1 -> 32 col tiles each
  const bf16* ar = sA + rowTile * 16 * 136;
  const v16bf a0 = loadA_bf(ar + 0, 136);
  const v16bf a1 = loadA_bf(ar + 32, 136);
  const v16bf a2 = loadA_bf(ar + 64, 136);
  const v16bf a3 = loadA_bf(ar + 96, 136);
  const int lane = tid & 31;
  const int lc = lane & 15;
  const int mb = (lane & 16) ? 8 : 0;
  for (int ct = cg * 32; ct < cg * 32 + 32; ++ct) {
    const bf16* br = ppb + (size_t)ct * 16 * HD;  // B^T rows = pp rows
    v8f acc = {};
    acc = wmma_bf16(a0, loadB_nk(br + 0, HD), acc);
    acc = wmma_bf16(a1, loadB_nk(br + 32, HD), acc);
    acc = wmma_bf16(a2, loadB_nk(br + 64, HD), acc);
    acc = wmma_bf16(a3, loadB_nk(br + 96, HD), acc);
    const int col = ct * 16 + lc;
#pragma unroll
    for (int v = 0; v < 8; ++v) {
      const float s = fast_sigmoid(acc[v]);
      const int r = rowTile * 16 + mb + v;
      pw[((size_t)b * LCV + r) * LPV + col] = (bf16)s;
      pwT[((size_t)b * LPV + col) * LCV + r] = (bf16)s;
    }
  }
}

// ---------------------------------------------------------------------------
// Comp side of diffusion step:
// c_tmp = tanh(ce@Whc0 + bhc0) * mc * (pw @ p_pre)
// logits_c[b,c] += c_tmp[b,c,:] . Whc1
// ---------------------------------------------------------------------------
__global__ __launch_bounds__(256) void k_ctmp(const float* __restrict__ ce,
                                              const float* __restrict__ Whc0,
                                              const float* __restrict__ bhc0,
                                              const float* __restrict__ mc,
                                              const bf16* __restrict__ ppre,
                                              const bf16* __restrict__ pw,
                                              const float* __restrict__ Whc1,
                                              float* __restrict__ logits_c) {
  const int b = blockIdx.x;
  __shared__ bf16 sW[HD * 136];   // Whc0 transposed
  __shared__ bf16 sCe[LCV * 136]; // ce[b] (bf16)
  __shared__ bf16 sP[HD * 40];    // p_pre chunk transposed: sP[n*40 + k], k<32
  const int tid = threadIdx.x;
  for (int idx = tid; idx < HD * HD; idx += 256) {
    const int k = idx >> 7, n = idx & 127;
    sW[n * 136 + k] = (bf16)Whc0[idx];
  }
  for (int idx = tid; idx < LCV * HD; idx += 256) {
    const int r = idx >> 7, k = idx & 127;
    sCe[r * 136 + k] = (bf16)ce[(size_t)b * LCV * HD + idx];
  }
  const int wave = tid >> 5;
  const int rowTile = wave & 3;
  const int cbase = (wave >> 2) * 4;
  const int lane = tid & 31;
  const int lc = lane & 15;
  const int mb = (lane & 16) ? 8 : 0;
  // hoisted uniform epilogue operands
  float bv[4], mv[4], w1[4];
#pragma unroll
  for (int t = 0; t < 4; ++t) {
    const int col = (cbase + t) * 16 + lc;
    bv[t] = bhc0[col];
    mv[t] = mc[b * HD + col];
    w1[t] = Whc1[col];
  }
  __syncthreads();
  const bf16* pwb = pw + (size_t)b * LCV * LPV + (size_t)rowTile * 16 * LPV;
  const bf16* ppb = ppre + (size_t)b * LPV * HD;
  v8f acc2[4] = {v8f{}, v8f{}, v8f{}, v8f{}};
  for (int kk = 0; kk < LPV; kk += 32) {
    __syncthreads();  // previous sP consumers done
    for (int idx = tid; idx < 32 * HD; idx += 256) {
      const int k = idx >> 7, n = idx & 127;
      sP[n * 40 + k] = ppb[(size_t)(kk + k) * HD + n];
    }
    __syncthreads();
    const v16bf a = loadA_bf(pwb + kk, LPV);
#pragma unroll
    for (int t = 0; t < 4; ++t) {
      const int ct = cbase + t;
      acc2[t] = wmma_bf16(a, loadB_nk(sP + ct * 16 * 40, 40), acc2[t]);
    }
  }
  // hidden projection accumulators + fused epilogue
  const bf16* ar = sCe + rowTile * 16 * 136;
  const v16bf c0 = loadA_bf(ar + 0, 136);
  const v16bf c1 = loadA_bf(ar + 32, 136);
  const v16bf c2 = loadA_bf(ar + 64, 136);
  const v16bf c3 = loadA_bf(ar + 96, 136);
  float rsum[8] = {0.f, 0.f, 0.f, 0.f, 0.f, 0.f, 0.f, 0.f};
#pragma unroll
  for (int t = 0; t < 4; ++t) {
    const int ct = cbase + t;
    const bf16* br = sW + ct * 16 * 136;
    v8f acc1 = {};
    acc1 = wmma_bf16(c0, loadB_nk(br + 0, 136), acc1);
    acc1 = wmma_bf16(c1, loadB_nk(br + 32, 136), acc1);
    acc1 = wmma_bf16(c2, loadB_nk(br + 64, 136), acc1);
    acc1 = wmma_bf16(c3, loadB_nk(br + 96, 136), acc1);
#pragma unroll
    for (int v = 0; v < 8; ++v) {
      const float val = fast_tanh(acc1[v] + bv[t]) * mv[t] * acc2[t][v];
      float part = val * w1[t];
      part += __shfl_xor(part, 1);
      part += __shfl_xor(part, 2);
      part += __shfl_xor(part, 4);
      part += __shfl_xor(part, 8);
      rsum[v] += part;
    }
  }
  if (lc == 0) {
#pragma unroll
    for (int v = 0; v < 8; ++v)
      atomicAdd(&logits_c[b * LCV + rowTile * 16 + mb + v], rsum[v]);
  }
}

// ---------------------------------------------------------------------------
// Prot side of diffusion step:
// p_tmp = tanh(pe@Whp0 + bhp0) * mp * (pw^T @ c_pre)
// logits_p[b,p] += p_tmp[b,p,:] . Whp1
// One block per (batch, 128-row slab of LP)
// ---------------------------------------------------------------------------
__global__ __launch_bounds__(256) void k_ptmp(const float* __restrict__ pe,
                                              const float* __restrict__ Whp0,
                                              const float* __restrict__ bhp0,
                                              const float* __restrict__ mp,
                                              const bf16* __restrict__ cpre,
                                              const bf16* __restrict__ pwT,
                                              const float* __restrict__ Whp1,
                                              float* __restrict__ logits_p) {
  const int b = blockIdx.x >> 3;
  const int slab = (blockIdx.x & 7) * 128;
  __shared__ bf16 sW[HD * 136];  // Whp0 transposed
  __shared__ bf16 sC[HD * 72];   // c_pre transposed: sC[n*72 + k], k<64
  const int tid = threadIdx.x;
  for (int idx = tid; idx < HD * HD; idx += 256) {
    const int k = idx >> 7, n = idx & 127;
    sW[n * 136 + k] = (bf16)Whp0[idx];
  }
  const bf16* cpb = cpre + (size_t)b * LCV * HD;
  for (int idx = tid; idx < LCV * HD; idx += 256) {
    const int k = idx >> 7, n = idx & 127;
    sC[n * 72 + k] = cpb[(size_t)k * HD + n];
  }
  __syncthreads();
  const int wave = tid >> 5;
  const int rowTile = wave;  // 8 row tiles of the 128-row slab
  const float* peb = pe + ((size_t)b * LPV + slab + rowTile * 16) * HD;
  const bf16* pwtb = pwT + ((size_t)b * LPV + slab + rowTile * 16) * LCV;
  const v16bf a0 = loadA_f32(peb + 0, HD);
  const v16bf a1 = loadA_f32(peb + 32, HD);
  const v16bf a2 = loadA_f32(peb + 64, HD);
  const v16bf a3 = loadA_f32(peb + 96, HD);
  const v16bf t0 = loadA_bf(pwtb + 0, LCV);
  const v16bf t1 = loadA_bf(pwtb + 32, LCV);
  const int lane = tid & 31;
  const int lc = lane & 15;
  const int mb = (lane & 16) ? 8 : 0;
  float rsum[8] = {0.f, 0.f, 0.f, 0.f, 0.f, 0.f, 0.f, 0.f};
#pragma unroll
  for (int ct = 0; ct < 8; ++ct) {
    const int col = ct * 16 + lc;
    const float bv = bhp0[col];
    const float mv = mp[b * HD + col];
    const float w1 = Whp1[col];
    const bf16* brW = sW + ct * 16 * 136;
    v8f acc1 = {};
    acc1 = wmma_bf16(a0, loadB_nk(brW + 0, 136), acc1);
    acc1 = wmma_bf16(a1, loadB_nk(brW + 32, 136), acc1);
    acc1 = wmma_bf16(a2, loadB_nk(brW + 64, 136), acc1);
    acc1 = wmma_bf16(a3, loadB_nk(brW + 96, 136), acc1);
    const bf16* brC = sC + ct * 16 * 72;
    v8f acc2 = {};
    acc2 = wmma_bf16(t0, loadB_nk(brC + 0, 72), acc2);
    acc2 = wmma_bf16(t1, loadB_nk(brC + 32, 72), acc2);
#pragma unroll
    for (int v = 0; v < 8; ++v) {
      const float val = fast_tanh(acc1[v] + bv) * mv * acc2[v];
      float part = val * w1;
      part += __shfl_xor(part, 1);
      part += __shfl_xor(part, 2);
      part += __shfl_xor(part, 4);
      part += __shfl_xor(part, 8);
      rsum[v] += part;
    }
  }
  if (lc == 0) {
#pragma unroll
    for (int v = 0; v < 8; ++v)
      atomicAdd(&logits_p[b * LPV + slab + rowTile * 16 + mb + v], rsum[v]);
  }
}

// ---------------------------------------------------------------------------
// Small kernels
// ---------------------------------------------------------------------------
__global__ void k_initlog(float* lc, float* lp, const float* bhc1,
                          const float* bhp1, int i) {
  const int t = blockIdx.x * blockDim.x + threadIdx.x;
  if (t < BATCH * LCV) lc[t] = bhc1[i];
  if (t < BATCH * LPV) lp[t] = bhp1[i];
}

__global__ __launch_bounds__(128) void k_mean(const float* __restrict__ ce,
                                              const float* __restrict__ pe,
                                              float* __restrict__ m) {
  const int b = blockIdx.x, h = threadIdx.x;
  float sc = 0.f, sp = 0.f;
  for (int c = 0; c < LCV; ++c) sc += ce[((size_t)b * LCV + c) * HD + h];
  for (int p = 0; p < LPV; ++p) sp += pe[((size_t)b * LPV + p) * HD + h];
  m[b * HD + h] = (sc * (1.f / LCV)) * (sp * (1.f / LPV));
}

__global__ __launch_bounds__(128) void k_attn_comp(const float* __restrict__ logits,
                                                   const float* __restrict__ ce,
                                                   float* __restrict__ cf) {
  const int b = blockIdx.x, tid = threadIdx.x;
  __shared__ float sl[LCV];
  __shared__ float red[128];
  __shared__ float ssum;
  float lm = -1e30f;
  if (tid < LCV) { const float v = logits[b * LCV + tid]; sl[tid] = v; lm = v; }
  red[tid] = lm;
  __syncthreads();
  for (int s = 64; s > 0; s >>= 1) {
    if (tid < s) red[tid] = fmaxf(red[tid], red[tid + s]);
    __syncthreads();
  }
  const float mx = red[0];
  __syncthreads();
  float ls = 0.f;
  if (tid < LCV) { const float e = __expf(sl[tid] - mx); sl[tid] = e; ls = e; }
  red[tid] = ls;
  __syncthreads();
  for (int s = 64; s > 0; s >>= 1) {
    if (tid < s) red[tid] += red[tid + s];
    __syncthreads();
  }
  if (tid == 0) ssum = red[0] + 1e-6f;
  __syncthreads();
  const float rinv = __builtin_amdgcn_rcpf(ssum);
  float acc = 0.f;
  const float* cb = ce + (size_t)b * LCV * HD + tid;
  for (int c = 0; c < LCV; ++c) acc += cb[(size_t)c * HD] * sl[c];
  cf[b * HD + tid] = acc * rinv;
}

__global__ __launch_bounds__(256) void k_attn_prot(const float* __restrict__ logits,
                                                   const float* __restrict__ pe,
                                                   float* __restrict__ pf) {
  const int b = blockIdx.x, tid = threadIdx.x;
  __shared__ float sl[LPV];
  __shared__ float red[256];
  __shared__ float ssum;
  float lm = -1e30f;
  for (int i = tid; i < LPV; i += 256) {
    const float v = logits[b * LPV + i];
    sl[i] = v;
    lm = fmaxf(lm, v);
  }
  red[tid] = lm;
  __syncthreads();
  for (int s = 128; s > 0; s >>= 1) {
    if (tid < s) red[tid] = fmaxf(red[tid], red[tid + s]);
    __syncthreads();
  }
  const float mx = red[0];
  __syncthreads();
  float ls = 0.f;
  for (int i = tid; i < LPV; i += 256) {
    const float e = __expf(sl[i] - mx);
    sl[i] = e;
    ls += e;
  }
  red[tid] = ls;
  __syncthreads();
  for (int s = 128; s > 0; s >>= 1) {
    if (tid < s) red[tid] += red[tid + s];
    __syncthreads();
  }
  if (tid == 0) ssum = red[0] + 1e-6f;
  __syncthreads();
  if (tid < HD) {
    const float rinv = __builtin_amdgcn_rcpf(ssum);
    float acc = 0.f;
    const float* pb = pe + (size_t)b * LPV * HD + tid;
    for (int p = 0; p < LPV; ++p) acc += pb[(size_t)p * HD] * sl[p];
    pf[b * HD + tid] = acc * rinv;
  }
}

// m_new = GRU(x=m, h=cf*pf)
__global__ __launch_bounds__(128) void k_gru(float* __restrict__ m,
                                             const float* __restrict__ cf,
                                             const float* __restrict__ pf,
                                             const float* __restrict__ Wih,
                                             const float* __restrict__ Whh,
                                             const float* __restrict__ bih,
                                             const float* __restrict__ bhh) {
  const int b = blockIdx.x, h = threadIdx.x;
  __shared__ float sx[HD], sh[HD];
  sx[h] = m[b * HD + h];
  sh[h] = cf[b * HD + h] * pf[b * HD + h];
  __syncthreads();
  float gi[3], gh[3];
#pragma unroll
  for (int g = 0; g < 3; ++g) {
    const float* wi = Wih + (size_t)(g * HD + h) * HD;
    const float* wh = Whh + (size_t)(g * HD + h) * HD;
    float a = 0.f, c = 0.f;
    for (int k = 0; k < HD; ++k) {
      a += sx[k] * wi[k];
      c += sh[k] * wh[k];
    }
    gi[g] = a + bih[g * HD + h];
    gh[g] = c + bhh[g * HD + h];
  }
  const float r = fast_sigmoid(gi[0] + gh[0]);
  const float z = fast_sigmoid(gi[1] + gh[1]);
  const float n = fast_tanh(gi[2] + r * gh[2]);
  m[b * HD + h] = (1.f - z) * n + z * sh[h];
}

// out[b] = sum_ij lrelu(cf[i]*pf[j]) * Wout[i*H+j] + b_out
__global__ __launch_bounds__(256) void k_out(const float* __restrict__ cf,
                                             const float* __restrict__ pf,
                                             const float* __restrict__ Wout,
                                             const float* __restrict__ bout,
                                             float* __restrict__ out) {
  const int b = blockIdx.x, tid = threadIdx.x;
  __shared__ float scf[HD], spf[HD], red[256];
  if (tid < HD) {
    scf[tid] = cf[b * HD + tid];
    spf[tid] = pf[b * HD + tid];
  }
  __syncthreads();
  float acc = 0.f;
  for (int idx = tid; idx < HD * HD; idx += 256) {
    const int i = idx >> 7, j = idx & 127;
    float k = scf[i] * spf[j];
    k = (k > 0.f) ? k : 0.1f * k;
    acc += k * Wout[idx];
  }
  red[tid] = acc;
  __syncthreads();
  for (int s = 128; s > 0; s >>= 1) {
    if (tid < s) red[tid] += red[tid + s];
    __syncthreads();
  }
  if (tid == 0) out[b] = red[0] + bout[0];
}

// ---------------------------------------------------------------------------
extern "C" void kernel_launch(void* const* d_in, const int* in_sizes, int n_in,
                              void* d_out, int out_size, void* d_ws, size_t ws_size,
                              hipStream_t stream) {
  const float* comp = (const float*)d_in[0];
  const float* prot = (const float*)d_in[1];
  const float* Wpc = (const float*)d_in[4];  const float* bpc = (const float*)d_in[5];
  const float* Wpp = (const float*)d_in[6];  const float* bpp = (const float*)d_in[7];
  const float* Wca = (const float*)d_in[8];  const float* bca = (const float*)d_in[9];
  const float* Wpa = (const float*)d_in[10]; const float* bpa = (const float*)d_in[11];
  const float* Wc2p = (const float*)d_in[12]; const float* bc2p = (const float*)d_in[13];
  const float* Wp2c = (const float*)d_in[14]; const float* bp2c = (const float*)d_in[15];
  const float* Wmc1 = (const float*)d_in[16]; const float* bmc1 = (const float*)d_in[17];
  const float* Wmp1 = (const float*)d_in[18]; const float* bmp1 = (const float*)d_in[19];
  const float* Whc0 = (const float*)d_in[20]; const float* bhc0 = (const float*)d_in[21];
  const float* Whp0 = (const float*)d_in[22]; const float* bhp0 = (const float*)d_in[23];
  const float* Whc1 = (const float*)d_in[24]; const float* bhc1 = (const float*)d_in[25];
  const float* Whp1 = (const float*)d_in[26]; const float* bhp1 = (const float*)d_in[27];
  const float* Wih = (const float*)d_in[28];  const float* Whh = (const float*)d_in[29];
  const float* bih = (const float*)d_in[30];  const float* bhh = (const float*)d_in[31];
  const float* Wout = (const float*)d_in[32]; const float* bout = (const float*)d_in[33];
  float* out = (float*)d_out;

  const size_t NC = (size_t)BATCH * LCV;   // 8192
  const size_t NP = (size_t)BATCH * LPV;   // 131072

  char* ws = (char*)d_ws;
  size_t off = 0;
  auto take = [&](size_t bytes) -> void* {
    void* p = ws + off;
    off += (bytes + 255) & ~(size_t)255;
    return p;
  };
  bf16* pc_bf   = (bf16*)take(NC * HD * sizeof(bf16));
  bf16* pp_bf   = (bf16*)take(NP * HD * sizeof(bf16));
  float* ce_f   = (float*)take(NC * HD * sizeof(float));
  float* pe_f   = (float*)take(NP * HD * sizeof(float));
  bf16* pw_bf   = (bf16*)take((size_t)BATCH * LCV * LPV * sizeof(bf16));
  bf16* pwT_bf  = (bf16*)take((size_t)BATCH * LPV * LCV * sizeof(bf16));
  bf16* cpre_bf = (bf16*)take(NC * HD * sizeof(bf16));
  bf16* ppre_bf = (bf16*)take(NP * HD * sizeof(bf16));
  float* m_f    = (float*)take((size_t)BATCH * HD * sizeof(float));
  float* mc_f   = (float*)take((size_t)BATCH * HD * sizeof(float));
  float* mp_f   = (float*)take((size_t)BATCH * HD * sizeof(float));
  float* logc_f = (float*)take((size_t)BATCH * LCV * sizeof(float));
  float* logp_f = (float*)take((size_t)BATCH * LPV * sizeof(float));
  float* cf_f   = (float*)take((size_t)BATCH * HD * sizeof(float));
  float* pf_f   = (float*)take((size_t)BATCH * HD * sizeof(float));
  (void)ws_size; (void)in_sizes; (void)n_in; (void)out_size;

  const int gC = (int)(NC / 64);   // 128
  const int gP = (int)(NP / 64);   // 2048
  const int gM = BATCH / 64;       // 2

  // Input projections (lrelu): pc/pp -> bf16, ce/pe -> f32
  k_proj<1, false, true><<<gC, 256, 0, stream>>>(comp, Wpc, bpc, nullptr, pc_bf);
  k_proj<1, false, true><<<gP, 256, 0, stream>>>(prot, Wpp, bpp, nullptr, pp_bf);
  k_proj<1, true, false><<<gC, 256, 0, stream>>>(comp, Wca, bca, ce_f, nullptr);
  k_proj<1, true, false><<<gP, 256, 0, stream>>>(prot, Wpa, bpa, pe_f, nullptr);

  // Pairwise interaction map
  k_pairwise<<<BATCH, 256, 0, stream>>>(pc_bf, pp_bf, pw_bf, pwT_bf);

  // m = mean(ce) * mean(pe)
  k_mean<<<BATCH, 128, 0, stream>>>(ce_f, pe_f, m_f);

  for (int i = 0; i < DLOOP; ++i) {
    const float* Wc2p_i = Wc2p + (size_t)i * HD * HD;
    const float* Wp2c_i = Wp2c + (size_t)i * HD * HD;
    const float* Whc0_i = Whc0 + (size_t)i * HD * HD;
    const float* Whp0_i = Whp0 + (size_t)i * HD * HD;
    const float* Wmc1_i = Wmc1 + (size_t)i * HD * HD;
    const float* Wmp1_i = Wmp1 + (size_t)i * HD * HD;

    // c_pre / p_pre (tanh -> bf16), mc / mp (tanh -> f32)
    k_proj<2, false, true><<<gC, 256, 0, stream>>>(ce_f, Wc2p_i, bc2p + i * HD, nullptr, cpre_bf);
    k_proj<2, false, true><<<gP, 256, 0, stream>>>(pe_f, Wp2c_i, bp2c + i * HD, nullptr, ppre_bf);
    k_proj<2, true, false><<<gM, 256, 0, stream>>>(m_f, Wmc1_i, bmc1 + i * HD, mc_f, nullptr);
    k_proj<2, true, false><<<gM, 256, 0, stream>>>(m_f, Wmp1_i, bmp1 + i * HD, mp_f, nullptr);

    // logits init with attention bias
    k_initlog<<<(BATCH * LPV + 255) / 256, 256, 0, stream>>>(logc_f, logp_f, bhc1, bhp1, i);

    // fused hidden-proj * gate * message GEMMs -> attention logits
    k_ctmp<<<BATCH, 256, 0, stream>>>(ce_f, Whc0_i, bhc0 + i * HD, mc_f, ppre_bf,
                                      pw_bf, Whc1 + i * HD, logc_f);
    k_ptmp<<<BATCH * 8, 256, 0, stream>>>(pe_f, Whp0_i, bhp0 + i * HD, mp_f, cpre_bf,
                                          pwT_bf, Whp1 + i * HD, logp_f);

    // segmented softmax + weighted sums
    k_attn_comp<<<BATCH, 128, 0, stream>>>(logc_f, ce_f, cf_f);
    k_attn_prot<<<BATCH, 256, 0, stream>>>(logp_f, pe_f, pf_f);

    // m = GRU(m, cf*pf)
    k_gru<<<BATCH, 128, 0, stream>>>(m_f, cf_f, pf_f, Wih, Whh, bih, bhh);
  }

  // output head
  k_out<<<BATCH, 256, 0, stream>>>(cf_f, pf_f, Wout, bout, out);
}
